// Grok1MoE_80238579024377
// MI455X (gfx1250) — matrix operand order repaired and checked by reading, not compiled, not run on hardware
//
#include <hip/hip_runtime.h>
#include <hip/hip_bf16.h>
#include <stdint.h>

// Problem constants (match reference)
#define T_TOK   2048
#define HID     1024
#define INTER   4096
#define NEXP    8
#define SOFTCAP_V 30.0f
#define TILE_M  32          // tokens per workgroup (2 M-subtiles of 16)

typedef __attribute__((ext_vector_type(16))) __bf16 v16bf;
typedef __attribute__((ext_vector_type(8)))  float  v8f;

union BF16x16 { uint16_t s[16]; uint4 q[2]; v16bf v; };

// round-to-nearest-even f32 -> bf16 (bit trick)
__device__ __forceinline__ uint16_t f2bf(float f){
  uint32_t u = __float_as_uint(f);
  u += 0x7FFFu + ((u >> 16) & 1u);
  return (uint16_t)(u >> 16);
}

// Load 16 consecutive f32 (16B-aligned) and convert to a bf16 fragment (slow path)
__device__ __forceinline__ v16bf cvt16(const float* __restrict__ p){
  BF16x16 b;
#pragma unroll
  for (int q = 0; q < 4; ++q){
    float4 f = reinterpret_cast<const float4*>(p)[q];
    b.s[q*4+0] = f2bf(f.x);
    b.s[q*4+1] = f2bf(f.y);
    b.s[q*4+2] = f2bf(f.z);
    b.s[q*4+3] = f2bf(f.w);
  }
  return b.v;
}

// B-fragment loader: bf16 direct (2x b128 loads) or f32 + inline convert
template<bool WBF>
__device__ __forceinline__ v16bf loadB(const float* __restrict__ pf,
                                       const uint16_t* __restrict__ pb){
  if constexpr (WBF){
    BF16x16 b;
    const uint4* q = reinterpret_cast<const uint4*>(pb);
    b.q[0] = q[0];
    b.q[1] = q[1];
    return b.v;
  } else {
    return cvt16(pf);
  }
}

// ---------------- fp32 -> bf16 conversion (grid-stride) ----------------
__global__ __launch_bounds__(256) void cvt_bf16_kernel(const float* __restrict__ src,
                                                       uint16_t* __restrict__ dst,
                                                       int n4){
  int stride = gridDim.x * 256;
  for (int i = blockIdx.x * 256 + threadIdx.x; i < n4; i += stride){
    float4 v = reinterpret_cast<const float4*>(src)[i];
    union { uint16_t s[4]; uint2 u; } r;
    r.s[0] = f2bf(v.x); r.s[1] = f2bf(v.y); r.s[2] = f2bf(v.z); r.s[3] = f2bf(v.w);
    reinterpret_cast<uint2*>(dst)[i] = r.u;
  }
}

// ---------------- router (softcap -> softmax -> top2 -> scatter) ----------------
__global__ __launch_bounds__(256) void router_kernel(const float* __restrict__ x,
                                                     const float* __restrict__ wg,
                                                     int* __restrict__ counts,
                                                     int* __restrict__ toklist,
                                                     int* __restrict__ pairslot,
                                                     float* __restrict__ pairw){
  int t    = blockIdx.x * 8 + (threadIdx.x >> 5);   // one wave32 per token
  int lane = threadIdx.x & 31;
  if (t >= T_TOK) return;

  float acc[NEXP];
#pragma unroll
  for (int e = 0; e < NEXP; ++e) acc[e] = 0.0f;

  const float* xr = x + (size_t)t * HID;
  for (int h = lane; h < HID; h += 32){
    float xv = xr[h];
#pragma unroll
    for (int e = 0; e < NEXP; ++e) acc[e] = fmaf(xv, wg[e * HID + h], acc[e]);
  }
#pragma unroll
  for (int e = 0; e < NEXP; ++e){
    float v = acc[e];
#pragma unroll
    for (int o = 16; o > 0; o >>= 1) v += __shfl_xor(v, o, 32);
    acc[e] = v;
  }
  if (lane == 0){
    float pr[NEXP]; float mx = -1e30f;
#pragma unroll
    for (int e = 0; e < NEXP; ++e){
      float l = SOFTCAP_V * tanhf(acc[e] / SOFTCAP_V);
      pr[e] = l; mx = fmaxf(mx, l);
    }
    float s = 0.0f;
#pragma unroll
    for (int e = 0; e < NEXP; ++e){ pr[e] = expf(pr[e] - mx); s += pr[e]; }
    float inv_s = 1.0f / s;
#pragma unroll
    for (int e = 0; e < NEXP; ++e) pr[e] *= inv_s;

    // top-2, lowest index wins ties (match jax.lax.top_k)
    int i1 = 0;
#pragma unroll
    for (int e = 1; e < NEXP; ++e) if (pr[e] > pr[i1]) i1 = e;
    int i2 = (i1 == 0) ? 1 : 0;
#pragma unroll
    for (int e = 0; e < NEXP; ++e) if (e != i1 && pr[e] > pr[i2]) i2 = e;

    float v1 = pr[i1], v2 = pr[i2];
    float invk = 1.0f / (v1 + v2);

    int p0 = atomicAdd(&counts[i1], 1);
    toklist[i1 * T_TOK + p0]  = t;
    pairslot[i1 * T_TOK + p0] = t * 2 + 0;
    pairw[i1 * T_TOK + p0]    = v1 * invk;

    int p1 = atomicAdd(&counts[i2], 1);
    toklist[i2 * T_TOK + p1]  = t;
    pairslot[i2 * T_TOK + p1] = t * 2 + 1;
    pairw[i2 * T_TOK + p1]    = v2 * invk;
  }
}

// ---------------- fused expert MLP (WMMA bf16, f32 accum) ----------------
// grid = (64 token tiles of 32, 8 experts), 256 threads = 8 waves.
// Wave w: M-subtile = w>>2 (16 tokens), quarter q = w&3 (I-tile in GEMM1,
// N-range [q*256, q*256+256) in GEMM2). 16 C-accumulator tiles per wave in VGPRs.
template<bool WBF>
__global__ __launch_bounds__(256) void moe_expert_kernel(
    const float* __restrict__ w1f, const float* __restrict__ w3f,
    const float* __restrict__ w2f,
    const uint16_t* __restrict__ w1b, const uint16_t* __restrict__ w3b,
    const uint16_t* __restrict__ w2b,
    const uint16_t* __restrict__ xbf, const int* __restrict__ counts,
    const int* __restrict__ toklist, const int* __restrict__ pairslot,
    const float* __restrict__ pairw, float* __restrict__ pairOut)
{
  const int e    = blockIdx.y;
  const int tile = blockIdx.x;
  const int cnt  = counts[e];
  if (tile * TILE_M >= cnt) return;

  __shared__ uint16_t xs[TILE_M * HID];    // 64 KB: 32 token rows of bf16 x
  __shared__ uint16_t acts[TILE_M * 64];   //  4 KB: act chunk [m][i_local]
  __shared__ int   s_tok[TILE_M];
  __shared__ int   s_slot[TILE_M];
  __shared__ float s_w[TILE_M];

  const int tid  = threadIdx.x;
  const int lane = tid & 31;
  const int wv   = tid >> 5;      // wave id 0..7
  const int ms   = wv >> 2;       // M-subtile 0/1
  const int wq   = wv & 3;        // quarter for I/N split
  const int mh   = lane >> 4;     // lane half (ISA fragment striping)
  const int lm   = lane & 15;

  if (tid < TILE_M){
    int r = tile * TILE_M + tid;
    bool valid = r < cnt;
    s_tok[tid]  = valid ? toklist[e * T_TOK + r]  : 0;
    s_slot[tid] = valid ? pairslot[e * T_TOK + r] : -1;
    s_w[tid]    = valid ? pairw[e * T_TOK + r]    : 0.0f;
  }
  __syncthreads();

  // stage 32 gathered token rows into LDS (uint4 = 8 bf16)
  for (int it = tid; it < (TILE_M * HID) / 8; it += 256){
    int row = it >> 7;            // HID/8 = 128 uint4 per row
    reinterpret_cast<uint4*>(xs)[it] =
        reinterpret_cast<const uint4*>(xbf + (size_t)s_tok[row] * HID)[it & 127];
  }
  __syncthreads();

  const size_t eoff = (size_t)e * INTER * HID;  // same extent for w1/w3/w2
  const float*    W1f = w1f + eoff;
  const float*    W3f = w3f + eoff;
  const float*    W2f = w2f + eoff;
  const uint16_t* W1b = w1b + eoff;
  const uint16_t* W3b = w3b + eoff;
  const uint16_t* W2b = w2b + eoff;

  const int arow = ms * 16 + lm;     // this wave's A row in xs/acts

  v8f acc[16];
#pragma unroll
  for (int n = 0; n < 16; ++n) acc[n] = (v8f){0,0,0,0,0,0,0,0};

  for (int ic = 0; ic < INTER; ic += 64){
    // ---- GEMM1: this wave's 16x16 act tile (I-tile = ic + wq*16), K = HID ----
    const int itile = ic + wq * 16;
    v8f c1 = (v8f){0,0,0,0,0,0,0,0};
    v8f c3 = (v8f){0,0,0,0,0,0,0,0};
    const size_t rowoff = (size_t)(itile + lm) * HID + mh * 16;
    for (int kc = 0; kc < HID; kc += 32){
      BF16x16 a;  // A fragment: row m=lm of subtile ms, contiguous k = 16*laneHalf + s
      const uint4* ap = reinterpret_cast<const uint4*>(&xs[arow * HID + kc + mh * 16]);
      a.q[0] = ap[0];
      a.q[1] = ap[1];   // +8 halves
      v16bf b1 = loadB<WBF>(W1f + rowoff + kc, W1b + rowoff + kc);
      v16bf b3 = loadB<WBF>(W3f + rowoff + kc, W3b + rowoff + kc);
      c1 = __builtin_amdgcn_wmma_f32_16x16x32_bf16(false, a.v, false, b1, (short)0, c1, false, false);
      c3 = __builtin_amdgcn_wmma_f32_16x16x32_bf16(false, a.v, false, b3, (short)0, c3, false, false);
    }
    // exact GELU(h1) * h3 -> bf16 act chunk in LDS
#pragma unroll
    for (int j = 0; j < 8; ++j){
      float g   = c1[j];
      float act = 0.5f * g * (1.0f + erff(g * 0.70710678118654752f)) * c3[j];
      int m = ms * 16 + j + 8 * mh;        // C layout: row = j + 8*laneHalf
      acts[m * 64 + wq * 16 + lm] = f2bf(act);
    }
    __syncthreads();

    // ---- GEMM2: out[16 x 256-per-wave] += act(16x64) @ w2^T chunk ----
    BF16x16 a0, a1;
    {
      const uint4* p0 = reinterpret_cast<const uint4*>(&acts[arow * 64 + mh * 16]);
      a0.q[0] = p0[0]; a0.q[1] = p0[1];
      const uint4* p1 = reinterpret_cast<const uint4*>(&acts[arow * 64 + 32 + mh * 16]);
      a1.q[0] = p1[0]; a1.q[1] = p1[1];
    }
#pragma unroll
    for (int nt = 0; nt < 16; ++nt){
      const int h = (wq * 16 + nt) * 16 + lm;
      const size_t w2off = (size_t)h * INTER + ic + mh * 16;
      v16bf b0 = loadB<WBF>(W2f + w2off, W2b + w2off);
      acc[nt] = __builtin_amdgcn_wmma_f32_16x16x32_bf16(false, a0.v, false, b0, (short)0, acc[nt], false, false);
      v16bf bx = loadB<WBF>(W2f + w2off + 32, W2b + w2off + 32);
      acc[nt] = __builtin_amdgcn_wmma_f32_16x16x32_bf16(false, a1.v, false, bx, (short)0, acc[nt], false, false);
    }
    __syncthreads();   // protect acts before next chunk overwrites it
  }

  // epilogue: scale by routing weight, write per-pair slot (no atomics)
#pragma unroll
  for (int nt = 0; nt < 16; ++nt){
    const int h = (wq * 16 + nt) * 16 + lm;
#pragma unroll
    for (int j = 0; j < 8; ++j){
      int m    = ms * 16 + j + 8 * mh;
      int slot = s_slot[m];
      if (slot >= 0)
        pairOut[(size_t)slot * HID + h] = acc[nt][j] * s_w[m];
    }
  }
}

// ---------------- combine the two expert contributions ----------------
__global__ __launch_bounds__(256) void combine_kernel(const float* __restrict__ pairOut,
                                                      float* __restrict__ out){
  int i = blockIdx.x * 256 + threadIdx.x;     // float4 index
  if (i >= T_TOK * HID / 4) return;
  int elem = i * 4;
  int t = elem >> 10;            // / HID
  int h = elem & (HID - 1);
  float4 a = *reinterpret_cast<const float4*>(pairOut + ((size_t)t * 2 + 0) * HID + h);
  float4 b = *reinterpret_cast<const float4*>(pairOut + ((size_t)t * 2 + 1) * HID + h);
  float4 r; r.x = a.x + b.x; r.y = a.y + b.y; r.z = a.z + b.z; r.w = a.w + b.w;
  *reinterpret_cast<float4*>(out + elem) = r;
}

extern "C" void kernel_launch(void* const* d_in, const int* in_sizes, int n_in,
                              void* d_out, int out_size, void* d_ws, size_t ws_size,
                              hipStream_t stream)
{
  const float* x  = (const float*)d_in[0];   // [1,2048,1024]
  const float* wg = (const float*)d_in[1];   // [8,1024]
  const float* w1 = (const float*)d_in[2];   // [8,4096,1024]
  const float* w3 = (const float*)d_in[3];   // [8,4096,1024]
  const float* w2 = (const float*)d_in[4];   // [8,1024,4096]
  float* out = (float*)d_out;

  const size_t WELEMS = (size_t)NEXP * INTER * HID;   // per weight tensor

  char* ws = (char*)d_ws;
  size_t off = 0;
  uint16_t* xbf   = (uint16_t*)(ws + off); off += (size_t)T_TOK * HID * 2;      // 4 MB
  int* counts     = (int*)(ws + off);      off += 256;
  int* toklist    = (int*)(ws + off);      off += (size_t)NEXP * T_TOK * 4;
  int* pairslot   = (int*)(ws + off);      off += (size_t)NEXP * T_TOK * 4;
  float* pairw    = (float*)(ws + off);    off += (size_t)NEXP * T_TOK * 4;
  float* pairOut  = (float*)(ws + off);    off += (size_t)T_TOK * 2 * HID * 4;  // 16.8 MB
  uint16_t* w1b   = (uint16_t*)(ws + off); off += WELEMS * 2;                   // 67.1 MB
  uint16_t* w3b   = (uint16_t*)(ws + off); off += WELEMS * 2;                   // 67.1 MB
  uint16_t* w2b   = (uint16_t*)(ws + off); off += WELEMS * 2;                   // 67.1 MB
  const bool fast = (ws_size >= off);      // deterministic per-harness choice

  hipMemsetAsync(counts, 0, NEXP * sizeof(int), stream);

  cvt_bf16_kernel<<<2048, 256, 0, stream>>>(x, xbf, T_TOK * HID / 4);
  router_kernel<<<T_TOK / 8, 256, 0, stream>>>(x, wg, counts, toklist, pairslot, pairw);

  if (fast){
    // pre-convert all expert weights to bf16 (bandwidth-bound, ~26us at HBM peak)
    const int n4 = (int)(WELEMS / 4);
    cvt_bf16_kernel<<<8192, 256, 0, stream>>>(w1, w1b, n4);
    cvt_bf16_kernel<<<8192, 256, 0, stream>>>(w3, w3b, n4);
    cvt_bf16_kernel<<<8192, 256, 0, stream>>>(w2, w2b, n4);
    moe_expert_kernel<true><<<dim3(T_TOK / TILE_M, NEXP), 256, 0, stream>>>(
        w1, w3, w2, w1b, w3b, w2b, xbf, counts, toklist, pairslot, pairw, pairOut);
  } else {
    moe_expert_kernel<false><<<dim3(T_TOK / TILE_M, NEXP), 256, 0, stream>>>(
        w1, w3, w2, w1b, w3b, w2b, xbf, counts, toklist, pairslot, pairw, pairOut);
  }

  combine_kernel<<<(T_TOK * HID / 4 + 255) / 256, 256, 0, stream>>>(pairOut, out);
}
